// EdgePredictor_57114475102685
// MI455X (gfx1250) — compile-verified, matrix operand order checked by reading, not run on hardware
//
#include <hip/hip_runtime.h>

typedef __attribute__((ext_vector_type(16))) __bf16 v16bf;
typedef __attribute__((ext_vector_type(8)))  __bf16 v8bf;
typedef __attribute__((ext_vector_type(2)))  __bf16 v2bf;
typedef __attribute__((ext_vector_type(8)))  float  v8f;

#define BM    64      // pairs per block
#define BK    64      // K-chunk staged in LDS (2 WMMA k-steps)
#define HDIM  256     // hidden dim / N

#if defined(__has_builtin)
#  if __has_builtin(__builtin_amdgcn_cvt_pk_bf16_f32)
#    define HAVE_CVT_PK_BF16 1
#  endif
#endif

// f32 -> bf16 round-to-nearest-even (raw bits) — fallback path
__device__ __forceinline__ unsigned short f2bf(float f) {
  unsigned u = __builtin_bit_cast(unsigned, f);
  u += 0x7FFFu + ((u >> 16) & 1u);
  return (unsigned short)(u >> 16);
}

// pack two f32 into one dword of bf16{lo,hi}
__device__ __forceinline__ unsigned pack2(float lo, float hi) {
#ifdef HAVE_CVT_PK_BF16
  v2bf v = __builtin_amdgcn_cvt_pk_bf16_f32(lo, hi);
  return __builtin_bit_cast(unsigned, v);
#else
  return (unsigned)f2bf(lo) | ((unsigned)f2bf(hi) << 16);
#endif
}

// Build a 16-element bf16 fragment from two contiguous 8-element LDS runs
// (matches the 16-bit A/B WMMA VGPR layout: K 8-runs split by lane>>4).
__device__ __forceinline__ v16bf frag_ld(const unsigned short* p) {
  v8bf a0 = *(const v8bf*)(p);
  v8bf a1 = *(const v8bf*)(p + 16);
  return __builtin_shufflevector(a0, a1, 0, 1, 2, 3, 4, 5, 6, 7,
                                          8, 9, 10, 11, 12, 13, 14, 15);
}

__global__ __launch_bounds__(256, 1)
void EdgePredictor_57114475102685_kernel(const float* __restrict__ h_nodes,
                                         const int*   __restrict__ dst,
                                         const int*   __restrict__ src,
                                         const float* __restrict__ W1,
                                         const float* __restrict__ W2,
                                         const float* __restrict__ W3,
                                         float* __restrict__ out) {
  __shared__ unsigned short Xs[2][BM * BK];     // 2 x  8 KB gathered input chunks (bf16)
  __shared__ unsigned short Ws[2][HDIM * BK];   // 2 x 32 KB weight chunks, transposed [n][k]
  __shared__ unsigned short As[BM * HDIM];      //     32 KB inter-layer activations (bf16)

  const int t     = threadIdx.x;
  const int lane  = t & 31;
  const int wave  = t >> 5;
  const int waveM = wave & 3;        // 4 waves across M
  const int waveN = wave >> 2;       // 2 waves across N (128 cols each)
  const int mBase = waveM * 16;
  const int lm    = lane & 15;       // row (A/C) or col (B) within tile
  const int khalf = lane >> 4;       // which K 8-run this lane holds

  const int pair0 = blockIdx.x * BM;

  // Xs staging mapping: each thread owns row xr, 16-float chunk xq
  const int xr = t >> 2;             // 0..63
  const int xq = (t & 3) * 16;       // 0,16,32,48
  const int nd = dst[pair0 + xr];
  const int ns = src[pair0 + xr];

  // ---- staging helpers (lambdas keep the double-buffer plumbing readable) ----
  auto stageX = [&](int s, int b) {  // gather one [64 x 64] chunk of concat(h[dst],h[src])
    const int kBase   = s * BK;
    const int node    = (kBase < 256) ? nd : ns;
    const int colBase = (kBase & 255) + xq;
    const float* rowp = h_nodes + (size_t)node * 256 + colBase;
#pragma unroll
    for (int i = 0; i < 4; ++i) {
      float4 f = ((const float4*)rowp)[i];
      unsigned* dsp = (unsigned*)&Xs[b][xr * BK + xq + i * 4];
      dsp[0] = pack2(f.x, f.y);
      dsp[1] = pack2(f.z, f.w);
    }
  };
  auto stageW = [&](const float* __restrict__ Wg, int s, int b) {
    const int kBase = s * BK;        // [kBase,kBase+64) x [0,256) -> Ws[b][n][k]
#pragma unroll
    for (int i = 0; i < 8; ++i) {
      int slot  = t + i * 256;       // 0..2047
      int cpair = slot >> 6;         // k = kBase + 2*cpair
      int n0    = (slot & 63) * 4;
      const float* base = Wg + (size_t)(kBase + 2 * cpair) * HDIM + n0;
      float4 fa = *(const float4*)(base);
      float4 fb = *(const float4*)(base + HDIM);
#pragma unroll
      for (int j = 0; j < 4; ++j) {
        *(unsigned*)&Ws[b][(n0 + j) * BK + 2 * cpair] = pack2((&fa.x)[j], (&fb.x)[j]);
      }
    }
  };
  auto compute = [&](const unsigned short* Abuf, int astride, int acol0,
                     const unsigned short* Wbuf, v8f* accp) {
#pragma unroll
    for (int kt = 0; kt < 2; ++kt) {
      const int ko = kt * 32 + khalf * 8;
      v16bf a = frag_ld(&Abuf[(mBase + lm) * astride + acol0 + ko]);
#pragma unroll
      for (int j = 0; j < 8; ++j) {
        const int n = waveN * 128 + j * 16 + lm;
        v16bf b = frag_ld(&Wbuf[n * BK + ko]);
        accp[j] = __builtin_amdgcn_wmma_f32_16x16x32_bf16(
            false, a, false, b, (short)0, accp[j], false, false);
      }
    }
  };
  auto storeAct = [&](v8f* accp) {   // relu -> bf16 -> As  (C layout: col lm, rows r+8*khalf)
#pragma unroll
    for (int j = 0; j < 8; ++j) {
      const int n = waveN * 128 + j * 16 + lm;
#pragma unroll
      for (int r = 0; r < 8; ++r) {
        float v = accp[j][r];
        v = v > 0.f ? v : 0.f;
        As[(mBase + khalf * 8 + r) * HDIM + n] = f2bf(v);
      }
    }
  };

  // ---------------- Layer 1:  X[64,512] @ W1[512,256], double-buffered ----------------
  v8f acc[8] = {};
  stageX(0, 0);
  stageW(W1, 0, 0);
  __syncthreads();
  for (int s = 0; s < 8; ++s) {
    const int cur = s & 1;
    compute(&Xs[cur][0], BK, 0, &Ws[cur][0], acc);
    if (s + 1 < 8) { stageX(s + 1, cur ^ 1); stageW(W1, s + 1, cur ^ 1); }
    __syncthreads();
  }

  storeAct(acc);                     // A1 -> LDS
  stageW(W2, 0, 0);                  // overlap first W2 stage with the A1 stores
  __syncthreads();

  // ---------------- Layer 2:  A1[64,256] @ W2[256,256], double-buffered ----------------
  v8f acc2[8] = {};
  for (int s = 0; s < 4; ++s) {
    const int cur = s & 1;
    compute(&As[0], HDIM, s * BK, &Ws[cur][0], acc2);
    if (s + 1 < 4) stageW(W2, s + 1, cur ^ 1);
    __syncthreads();
  }

  storeAct(acc2);                    // A2 -> LDS (reuse; all A1 reads fenced above)
  __syncthreads();

  // ---------------- Layer 3:  A2[64,256] @ W3[256,2] ----------------
  if (t < BM * 2) {
    const int row = t >> 1;
    const int col = t & 1;
    float sum = 0.f;
#pragma unroll 8
    for (int k = 0; k < HDIM; ++k) {
      float a = __builtin_bit_cast(float, (unsigned)As[row * HDIM + k] << 16);
      sum += a * W3[k * 2 + col];
    }
    out[(size_t)(pair0 + row) * 2 + col] = sum;
  }
}

extern "C" void kernel_launch(void* const* d_in, const int* in_sizes, int n_in,
                              void* d_out, int out_size, void* d_ws, size_t ws_size,
                              hipStream_t stream) {
  const float* h_nodes = (const float*)d_in[0];
  const int*   dst     = (const int*)d_in[1];
  const int*   src     = (const int*)d_in[2];
  const float* W1      = (const float*)d_in[3];
  const float* W2      = (const float*)d_in[4];
  const float* W3      = (const float*)d_in[5];
  float*       out     = (float*)d_out;

  const int P = in_sizes[1];          // 524288 pairs
  const int nblocks = P / BM;         // 8192
  EdgePredictor_57114475102685_kernel<<<nblocks, 256, 0, stream>>>(
      h_nodes, dst, src, W1, W2, W3, out);
}
